// TransformerEncoderLayer_83056077570746
// MI455X (gfx1250) — matrix operand order repaired and checked by reading, not compiled, not run on hardware
//
#include <hip/hip_runtime.h>
#include <hip/hip_bf16.h>

#define D_MODEL 768
#define N_HEADS 12
#define D_HEAD  64
#define D_FF    3072
#define T_SEQ   4096
#define LN_EPS  1e-5f

typedef __bf16 bf16;
typedef __attribute__((ext_vector_type(16))) __bf16 v16bf;
typedef __attribute__((ext_vector_type(8)))  __bf16 v8bf;
typedef __attribute__((ext_vector_type(8)))  float  v8f;

union V16U { v16bf v; v8bf h[2]; };

// A-fragment (16-bit A 16x32): lane&15 -> row, hi half-wave offsets K by 16.
// elements 0..15 <- K = base+0..15
__device__ inline v16bf load_frag_split(const bf16* p) {
    V16U u;
    u.h[0] = *(const v8bf*)(p);
    u.h[1] = *(const v8bf*)(p + 8);
    return u.v;
}

// B-fragment (16-bit B 32x16): lane&15 -> column, hi half-wave owns K=16..31.
__device__ inline v16bf load_frag_contig(const bf16* p) {
    V16U u;
    u.h[0] = ((const v8bf*)p)[0];
    u.h[1] = ((const v8bf*)p)[1];
    return u.v;
}

__device__ inline v8f wmma_bf16(v16bf a, v16bf b, v8f c) {
    return __builtin_amdgcn_wmma_f32_16x16x32_bf16(false, a, false, b, (short)0, c,
                                                   false, false);
}

// ---------------------------------------------------------------------------
// f32 -> bf16 elementwise convert
// ---------------------------------------------------------------------------
__global__ void cvt_f32_bf16_kernel(const float* __restrict__ src,
                                    bf16* __restrict__ dst, int n) {
    int i = blockIdx.x * 256 + threadIdx.x;
    if (i < n) dst[i] = (bf16)src[i];
}

// ---------------------------------------------------------------------------
// LayerNorm over D=768: one 256-thread block per token row; bf16 output.
// ---------------------------------------------------------------------------
__global__ void layernorm_kernel(const float* __restrict__ X,
                                 const float* __restrict__ gamma,
                                 const float* __restrict__ beta,
                                 bf16* __restrict__ H) {
    __shared__ float red[256];
    const int tid = threadIdx.x;
    const size_t row = blockIdx.x;
    const float* xr = X + row * D_MODEL;
    float v0 = xr[tid], v1 = xr[tid + 256], v2 = xr[tid + 512];

    red[tid] = v0 + v1 + v2;
    __syncthreads();
    for (int o = 128; o > 0; o >>= 1) {
        if (tid < o) red[tid] += red[tid + o];
        __syncthreads();
    }
    const float mean = red[0] * (1.0f / D_MODEL);
    __syncthreads();

    float d0 = v0 - mean, d1 = v1 - mean, d2 = v2 - mean;
    red[tid] = d0 * d0 + d1 * d1 + d2 * d2;
    __syncthreads();
    for (int o = 128; o > 0; o >>= 1) {
        if (tid < o) red[tid] += red[tid + o];
        __syncthreads();
    }
    const float rstd = rsqrtf(red[0] * (1.0f / D_MODEL) + LN_EPS);

    bf16* hr = H + row * D_MODEL;
    hr[tid]       = (bf16)(gamma[tid]       * d0 * rstd + beta[tid]);
    hr[tid + 256] = (bf16)(gamma[tid + 256] * d1 * rstd + beta[tid + 256]);
    hr[tid + 512] = (bf16)(gamma[tid + 512] * d2 * rstd + beta[tid + 512]);
}

// ---------------------------------------------------------------------------
// WMMA GEMM: C[M,N] = A[M,K] (bf16) x W[N,K]^T (bf16)
//   mode 0: bf16 C[m*N + n]
//   mode 1: bf16 transposed C[n*M + m]         (used to build V^T)
//   mode 2: f32  C = acc + bias[n] + resid[m*N+n]
//   mode 3: bf16 C = gelu(acc + bias[n])       (exact erf GELU)
// Wave tile 32x32 (2x2 WMMA register block: 2 A-frags + 2 B-frags -> 4 WMMA).
// Block: 256 threads (8 waves) arranged 2(M) x 4(N): block tile 64 x 128.
// Grid: (N/128, M/64).
// ---------------------------------------------------------------------------
__global__ __launch_bounds__(256)
void gemm_bf16_kernel(const bf16* __restrict__ A, const bf16* __restrict__ W,
                      int Mdim, int Ndim, int Kdim,
                      void* __restrict__ Cout, int mode,
                      const float* __restrict__ bias,
                      const float* __restrict__ resid) {
    const int lane = threadIdx.x & 31;
    const int w    = threadIdx.x >> 5;
    const int ln   = lane & 15;
    const int hi   = lane >> 4;

    const int m0 = blockIdx.y * 64  + (w & 1) * 32;
    const int n0 = blockIdx.x * 128 + (w >> 1) * 32;

    const bf16* ap0 = A + (size_t)(m0 + ln)      * Kdim + hi * 16;
    const bf16* ap1 = A + (size_t)(m0 + 16 + ln) * Kdim + hi * 16;
    const bf16* bp0 = W + (size_t)(n0 + ln)      * Kdim + hi * 16;
    const bf16* bp1 = W + (size_t)(n0 + 16 + ln) * Kdim + hi * 16;

    v8f acc00 = {}, acc01 = {}, acc10 = {}, acc11 = {};
    for (int k0 = 0; k0 < Kdim; k0 += 32) {
        const v16bf a0 = load_frag_split(ap0 + k0);
        const v16bf a1 = load_frag_split(ap1 + k0);
        const v16bf b0 = load_frag_contig(bp0 + k0);
        const v16bf b1 = load_frag_contig(bp1 + k0);
        acc00 = wmma_bf16(a0, b0, acc00);
        acc01 = wmma_bf16(a0, b1, acc01);
        acc10 = wmma_bf16(a1, b0, acc10);
        acc11 = wmma_bf16(a1, b1, acc11);
    }

    v8f accs[2][2] = {{acc00, acc01}, {acc10, acc11}};
#pragma unroll
    for (int tm = 0; tm < 2; ++tm) {
#pragma unroll
        for (int tn = 0; tn < 2; ++tn) {
            const v8f acc   = accs[tm][tn];
            const int ncol  = n0 + tn * 16 + ln;
            const int mbase = m0 + tm * 16 + hi * 8;
            if (mode == 0) {
                bf16* C = (bf16*)Cout;
#pragma unroll
                for (int i = 0; i < 8; ++i)
                    C[(size_t)(mbase + i) * Ndim + ncol] = (bf16)acc[i];
            } else if (mode == 1) {
                bf16* C = (bf16*)Cout;  // transposed: C[n*M + m]
#pragma unroll
                for (int i = 0; i < 8; ++i)
                    C[(size_t)ncol * Mdim + (mbase + i)] = (bf16)acc[i];
            } else if (mode == 2) {
                float* C = (float*)Cout;
                const float bn = bias[ncol];
#pragma unroll
                for (int i = 0; i < 8; ++i) {
                    const size_t idx = (size_t)(mbase + i) * Ndim + ncol;
                    C[idx] = acc[i] + bn + resid[idx];
                }
            } else {  // mode 3: bias + exact GELU -> bf16
                bf16* C = (bf16*)Cout;
                const float bn = bias[ncol];
#pragma unroll
                for (int i = 0; i < 8; ++i) {
                    const float x = acc[i] + bn;
                    const float g =
                        0.5f * x * (1.0f + erff(x * 0.70710678118654752f));
                    C[(size_t)(mbase + i) * Ndim + ncol] = (bf16)g;
                }
            }
        }
    }
}

// ---------------------------------------------------------------------------
// Flash attention: one wave per (head, 32-query tile); stream 32-key tiles.
// K/V fragments are reused across both 16-query sub-tiles (2x operand reuse).
// Q,K in (T, D) bf16 row-major; Vt in (D, T) bf16 (transposed); Out (T, D) bf16.
// Block: 128 threads (4 waves). Grid: H * (T/32) / 4 = 384.
// ---------------------------------------------------------------------------
__global__ __launch_bounds__(128)
void attention_kernel(const bf16* __restrict__ Q, const bf16* __restrict__ Kb,
                      const bf16* __restrict__ Vt, bf16* __restrict__ Out) {
    __shared__ __align__(16) bf16 pT[4][2][16 * 32];

    const int lane = threadIdx.x & 31;
    const int w    = threadIdx.x >> 5;
    const int ln   = lane & 15;
    const int hi   = lane >> 4;

    const int gw   = blockIdx.x * 4 + w;
    const int head = gw >> 7;        // T/32 = 128 query tiles per head
    const int q0   = (gw & 127) * 32;
    const float scale = 0.125f;      // 1/sqrt(64)

    // Q A-fragments: [qtile][d-slab], kept in VGPRs for the whole key loop
    v16bf qa[2][2];
#pragma unroll
    for (int t = 0; t < 2; ++t) {
        const bf16* qp =
            Q + (size_t)(q0 + 16 * t + ln) * D_MODEL + head * D_HEAD + hi * 16;
        qa[t][0] = load_frag_split(qp);
        qa[t][1] = load_frag_split(qp + 32);
    }

    v8f o[2][4] = {};
    float mrow[2][8], lrow[2][8];
#pragma unroll
    for (int t = 0; t < 2; ++t)
#pragma unroll
        for (int i = 0; i < 8; ++i) { mrow[t][i] = -1e30f; lrow[t][i] = 0.0f; }

    for (int kt = 0; kt < T_SEQ; kt += 32) {
        // ----- K B-fragments (shared by both query tiles) -----
        const bf16* kp0 =
            Kb + (size_t)(kt + ln) * D_MODEL + head * D_HEAD + hi * 16;
        const bf16* kp1 = kp0 + 16 * (size_t)D_MODEL;
        const v16bf kb00 = load_frag_contig(kp0);
        const v16bf kb01 = load_frag_contig(kp0 + 32);
        const v16bf kb10 = load_frag_contig(kp1);
        const v16bf kb11 = load_frag_contig(kp1 + 32);

        // ----- S = Q K^T : 8 WMMAs -----
        v8f s[2][2];
#pragma unroll
        for (int t = 0; t < 2; ++t) {
            v8f z0 = {}, z1 = {};
            z0 = wmma_bf16(qa[t][0], kb00, z0);
            z0 = wmma_bf16(qa[t][1], kb01, z0);
            z1 = wmma_bf16(qa[t][0], kb10, z1);
            z1 = wmma_bf16(qa[t][1], kb11, z1);
            s[t][0] = z0; s[t][1] = z1;
        }

        // ----- online softmax per query tile -----
#pragma unroll
        for (int t = 0; t < 2; ++t) {
            float tmax[8];
#pragma unroll
            for (int i = 0; i < 8; ++i) tmax[i] = fmaxf(s[t][0][i], s[t][1][i]);
#pragma unroll
            for (int msk = 1; msk < 16; msk <<= 1)
#pragma unroll
                for (int i = 0; i < 8; ++i)
                    tmax[i] = fmaxf(tmax[i], __shfl_xor(tmax[i], msk, 32));

            float p0[8], p1[8], rs[8];
#pragma unroll
            for (int i = 0; i < 8; ++i) {
                const float mnew = fmaxf(mrow[t][i], tmax[i] * scale);
                const float alpha = __expf(mrow[t][i] - mnew);
                mrow[t][i] = mnew;
                p0[i] = __expf(s[t][0][i] * scale - mnew);
                p1[i] = __expf(s[t][1][i] * scale - mnew);
                rs[i] = p0[i] + p1[i];
                lrow[t][i] *= alpha;
#pragma unroll
                for (int n = 0; n < 4; ++n) o[t][n][i] *= alpha;
            }
#pragma unroll
            for (int msk = 1; msk < 16; msk <<= 1)
#pragma unroll
                for (int i = 0; i < 8; ++i)
                    rs[i] += __shfl_xor(rs[i], msk, 32);
#pragma unroll
            for (int i = 0; i < 8; ++i) lrow[t][i] += rs[i];

            // stage P tile (16x32 bf16) through LDS for A-fragment layout
            bf16* myP = pT[w][t];
#pragma unroll
            for (int i = 0; i < 8; ++i) {
                const int r = i + hi * 8;
                myP[r * 32 + ln]      = (bf16)p0[i];
                myP[r * 32 + 16 + ln] = (bf16)p1[i];
            }
        }
        __builtin_amdgcn_wave_barrier();
        asm volatile("s_wait_dscnt 0" ::: "memory");
        v16bf pa[2];
        pa[0] = load_frag_split(pT[w][0] + ln * 32 + hi * 16);
        pa[1] = load_frag_split(pT[w][1] + ln * 32 + hi * 16);
        __builtin_amdgcn_wave_barrier();

        // ----- O += P V : V B-fragments shared by both query tiles, 8 WMMAs --
        const bf16* vp =
            Vt + (size_t)(head * D_HEAD + ln) * T_SEQ + kt + hi * 16;
        v16bf vb[4];
#pragma unroll
        for (int n = 0; n < 4; ++n)
            vb[n] = load_frag_contig(vp + (size_t)(16 * n) * T_SEQ);
#pragma unroll
        for (int t = 0; t < 2; ++t)
#pragma unroll
            for (int n = 0; n < 4; ++n)
                o[t][n] = wmma_bf16(pa[t], vb[n], o[t][n]);
    }

    // ----- normalize and store (T, D) bf16 -----
#pragma unroll
    for (int t = 0; t < 2; ++t)
#pragma unroll
        for (int i = 0; i < 8; ++i) {
            const float inv = 1.0f / lrow[t][i];
            const size_t m = q0 + 16 * t + i + 8 * hi;
            bf16* orow = Out + m * D_MODEL + head * D_HEAD + ln;
            orow[0]  = (bf16)(o[t][0][i] * inv);
            orow[16] = (bf16)(o[t][1][i] * inv);
            orow[32] = (bf16)(o[t][2][i] * inv);
            orow[48] = (bf16)(o[t][3][i] * inv);
        }
}

// ---------------------------------------------------------------------------
// Host-side orchestration
// ---------------------------------------------------------------------------
extern "C" void kernel_launch(void* const* d_in, const int* in_sizes, int n_in,
                              void* d_out, int out_size, void* d_ws, size_t ws_size,
                              hipStream_t stream) {
    const float* x      = (const float*)d_in[0];
    const float* wq     = (const float*)d_in[1];
    const float* wk     = (const float*)d_in[2];
    const float* wv     = (const float*)d_in[3];
    const float* wo     = (const float*)d_in[4];
    const float* bo     = (const float*)d_in[5];
    const float* gamma1 = (const float*)d_in[6];
    const float* beta1  = (const float*)d_in[7];
    const float* gamma2 = (const float*)d_in[8];
    const float* beta2  = (const float*)d_in[9];
    const float* w1     = (const float*)d_in[10];
    const float* b1     = (const float*)d_in[11];
    const float* w2     = (const float*)d_in[12];
    const float* b2     = (const float*)d_in[13];
    float* out = (float*)d_out;

    // ---- workspace layout (bf16 unless noted) ----
    char* p = (char*)d_ws;
    const size_t TD  = (size_t)T_SEQ * D_MODEL;
    const size_t TF  = (size_t)T_SEQ * D_FF;
    const size_t DD  = (size_t)D_MODEL * D_MODEL;
    const size_t DFD = (size_t)D_FF * D_MODEL;

    bf16* hbf    = (bf16*)p;            p += TD * 2;
    bf16* qbf    = (bf16*)p;            p += TD * 2;
    bf16* kbf    = (bf16*)p;            p += TD * 2;
    bf16* vtbf   = (bf16*)p;            p += TD * 2;   // V^T : (D, T)
    bf16* attnbf = (bf16*)p;            p += TD * 2;
    float* x1f   = (float*)p;           p += TD * 4;   // residual stream post-attn
    bf16* h2bf   = (bf16*)p;            p += TD * 2;
    bf16* ff1bf  = (bf16*)p;            p += TF * 2;
    bf16* wqb    = (bf16*)p;            p += DD * 2;
    bf16* wkb    = (bf16*)p;            p += DD * 2;
    bf16* wvb    = (bf16*)p;            p += DD * 2;
    bf16* wob    = (bf16*)p;            p += DD * 2;
    bf16* w1b    = (bf16*)p;            p += DFD * 2;
    bf16* w2b    = (bf16*)p;            p += DFD * 2;
    (void)ws_size; (void)n_in; (void)in_sizes; (void)out_size;

    // ---- 1) weights -> bf16 ----
    auto cvt = [&](const float* s, bf16* d, size_t n) {
        cvt_f32_bf16_kernel<<<dim3((n + 255) / 256), dim3(256), 0, stream>>>(s, d, (int)n);
    };
    cvt(wq, wqb, DD);  cvt(wk, wkb, DD);  cvt(wv, wvb, DD);  cvt(wo, wob, DD);
    cvt(w1, w1b, DFD); cvt(w2, w2b, DFD);

    // ---- 2) h = LN1(x) ----
    layernorm_kernel<<<dim3(T_SEQ), dim3(256), 0, stream>>>(x, gamma1, beta1, hbf);

    // ---- 3) Q, K, V projections (V written transposed) ----
    const dim3 gblk(256);
    const dim3 gDD(D_MODEL / 128, T_SEQ / 64);
    gemm_bf16_kernel<<<gDD, gblk, 0, stream>>>(hbf, wqb, T_SEQ, D_MODEL, D_MODEL,
                                               qbf, 0, nullptr, nullptr);
    gemm_bf16_kernel<<<gDD, gblk, 0, stream>>>(hbf, wkb, T_SEQ, D_MODEL, D_MODEL,
                                               kbf, 0, nullptr, nullptr);
    gemm_bf16_kernel<<<gDD, gblk, 0, stream>>>(hbf, wvb, T_SEQ, D_MODEL, D_MODEL,
                                               vtbf, 1, nullptr, nullptr);

    // ---- 4) flash attention ----
    attention_kernel<<<dim3(N_HEADS * (T_SEQ / 32) / 4), dim3(128), 0, stream>>>(
        qbf, kbf, vtbf, attnbf);

    // ---- 5) x1 = x + attn @ wo^T + bo ----
    gemm_bf16_kernel<<<gDD, gblk, 0, stream>>>(attnbf, wob, T_SEQ, D_MODEL, D_MODEL,
                                               x1f, 2, bo, x);

    // ---- 6) h2 = LN2(x1) ----
    layernorm_kernel<<<dim3(T_SEQ), dim3(256), 0, stream>>>(x1f, gamma2, beta2, h2bf);

    // ---- 7) ff1 = gelu(h2 @ w1^T + b1) ----
    gemm_bf16_kernel<<<dim3(D_FF / 128, T_SEQ / 64), gblk, 0, stream>>>(
        h2bf, w1b, T_SEQ, D_FF, D_MODEL, ff1bf, 3, b1, nullptr);

    // ---- 8) out = x1 + ff1 @ w2^T + b2 ----
    gemm_bf16_kernel<<<gDD, gblk, 0, stream>>>(ff1bf, w2b, T_SEQ, D_MODEL, D_FF,
                                               out, 2, b2, x1f);
}
